// EnhancedFrequencyLogGaborConv2d_70626442215906
// MI455X (gfx1250) — matrix operand order repaired and checked by reading, not compiled, not run on hardware
//
#include <hip/hip_runtime.h>
#include <math.h>

// ---------------------------------------------------------------------------
// EnhancedFrequencyLogGaborConv2d for MI455X (gfx1250, wave32, WMMA).
// - DFT/iDFT as 192^3 GEMMs on v_wmma_f32_16x16x32_f16 (f32 accumulate);
//   fftshift folded into twiddle signs, batch-invariant iFFT (64 imgs not 256)
// - all WMMA operand buffers stored f16 (convert once on store, not 12x on
//   load); B operands stored transposed so every fragment load is contiguous
// - attention fused per-pixel (one wave/pixel): h2/attn never materialized
// - global_prefetch of next k-step operands in the GEMM loops
// ---------------------------------------------------------------------------

#define HH   192
#define WW   192
#define HW   (192 * 192)
#define BB   4
#define CINC 3
#define COUTC 64
#define NCH  192          // COUT*NSC
#define HID  64
#define NPIX (BB * HW)
#define PI_F 3.14159265358979323846f

typedef __attribute__((ext_vector_type(16))) _Float16 v16h;
typedef __attribute__((ext_vector_type(8)))  float    v8f;

// ------------------------------ WMMA helpers -------------------------------

__device__ __forceinline__ v8f zero8() {
  v8f z;
#pragma unroll
  for (int i = 0; i < 8; ++i) z[i] = 0.0f;
  return z;
}

__device__ __forceinline__ v8f wmma16(v16h a, v16h b, v8f c) {
  // (neg_a, A, neg_b, B, c_mod, C, reuse_a, reuse_b)
  return __builtin_amdgcn_wmma_f32_16x16x32_f16(false, a, false, b, (short)0,
                                                c, false, false);
}

// A fragment: 16x32, f16 row-major source. lane<16 holds K 0..15, lane>=16 K 16..31.
__device__ __forceinline__ v16h fragAh(const _Float16* __restrict__ A, int lda,
                                       int r0, int k0, int lane) {
  int h = (lane >> 4) & 1;
  int m = lane & 15;
  const _Float16* row = A + (size_t)(r0 + m) * lda + k0 + (h << 4);
  v16h a;
#pragma unroll
  for (int e = 0; e < 16; ++e)
    a[e] = row[e];
  return a;
}

// B fragment from transposed f16 storage: element(k,n) = B[(n0+n)*ldb + k0+k]
// -> 16 contiguous f16 per lane (vectorizable).
__device__ __forceinline__ v16h fragBTh(const _Float16* __restrict__ B, int ldb,
                                        int k0, int n0, int lane, bool neg) {
  int h = (lane >> 4) & 1;
  int n = lane & 15;
  const _Float16* p = B + (size_t)(n0 + n) * ldb + k0 + (h << 4);
  v16h b;
#pragma unroll
  for (int e = 0; e < 16; ++e) {
    _Float16 v = p[e];
    b[e] = neg ? (_Float16)(-v) : v;
  }
  return b;
}

// f32 C/D tile store: VGPR j -> row r0 + j + 8*(lane>=16), col n0 + (lane&15)
__device__ __forceinline__ void storeTile(float* __restrict__ C, int ldc,
                                          int r0, int n0, int lane, v8f acc) {
  int h = (lane >> 4) & 1;
  int n = lane & 15;
#pragma unroll
  for (int j = 0; j < 8; ++j)
    C[(size_t)(r0 + j + h * 8) * ldc + n0 + n] = acc[j];
}

// f16 tile store, natural orientation (for buffers later used as A operands)
__device__ __forceinline__ void storeTileH(_Float16* __restrict__ C, int ldc,
                                           int r0, int n0, int lane, v8f acc) {
  int h = (lane >> 4) & 1;
  int n = lane & 15;
#pragma unroll
  for (int j = 0; j < 8; ++j)
    C[(size_t)(r0 + j + h * 8) * ldc + n0 + n] = (_Float16)acc[j];
}

// f16 tile store, transposed (for buffers later used as B operands):
// element(row,col) -> C[col*ldc + row]; 8 contiguous f16 per lane.
__device__ __forceinline__ void storeTileHT(_Float16* __restrict__ C, int ldc,
                                            int r0, int n0, int lane, v8f acc) {
  int h = (lane >> 4) & 1;
  int n = lane & 15;
  _Float16* p = C + (size_t)(n0 + n) * ldc + r0 + h * 8;
#pragma unroll
  for (int j = 0; j < 8; ++j) p[j] = (_Float16)acc[j];
}

// ---------------------------- wave reductions ------------------------------

__device__ __forceinline__ float waveSum(float v) {
#pragma unroll
  for (int m = 16; m >= 1; m >>= 1) v += __shfl_xor(v, m, 32);
  return v;
}
__device__ __forceinline__ float waveMax(float v) {
#pragma unroll
  for (int m = 16; m >= 1; m >>= 1) v = fmaxf(v, __shfl_xor(v, m, 32));
  return v;
}

// ------------------------------- kernels -----------------------------------

// Twiddle tables (f16, fftshift folded: (-1)^col forward, (-1)^row inverse,
// inverse UNSCALED - the 1/192^2 moves to buildq so f16 range is safe),
// x and w2 pre-converted to f16, band indices, BN stats zeroed.
__global__ void flg_tables_kernel(const float* __restrict__ fbs,
                                  const float* __restrict__ x,
                                  const float* __restrict__ w2,
                                  _Float16* __restrict__ Fsr, _Float16* __restrict__ Fsi,
                                  _Float16* __restrict__ IFsr, _Float16* __restrict__ IFsi,
                                  _Float16* __restrict__ xh, _Float16* __restrict__ w2h,
                                  int* __restrict__ idxlo, int* __restrict__ idxhi,
                                  float* __restrict__ stats /*384*/) {
  int t = blockIdx.x * 256 + threadIdx.x;
  if (t < 384) stats[t] = 0.0f;
  if (t < COUTC * CINC) {
    idxlo[t] = (int)floorf((fbs[t * 2 + 0] + 1.0f) * 0.5f * (float)HH);
    idxhi[t] = (int)floorf((fbs[t * 2 + 1] + 1.0f) * 0.5f * (float)HH);
  }
  if (t >= HW) return;
#pragma unroll
  for (int j = 0; j < 12; ++j)                 // 12 input images -> f16
    xh[(size_t)j * HW + t] = (_Float16)x[(size_t)j * HW + t];
  if (t < NCH * HID) w2h[t] = (_Float16)w2[t];
  int row = t / WW, col = t % WW;
  int m = (row * col) % HH;                    // exact angle reduction
  float ang = 6.283185307179586f * (float)m / 192.0f;
  float c = cosf(ang), s = sinf(ang);
  float sgn_col = (col & 1) ? -1.0f : 1.0f;    // forward: (-1)^input-index
  Fsr[t] = (_Float16)(sgn_col * c);            // cos(-ang)
  Fsi[t] = (_Float16)(sgn_col * (-s));         // sin(-ang)
  float sgn_row = (row & 1) ? -1.0f : 1.0f;    // inverse: (-1)^output-index
  IFsr[t] = (_Float16)(sgn_row * c);           // unscaled (+2pi rotation)
  IFsi[t] = (_Float16)(sgn_row * s);
}

// T^T[img] = (x[img] * Fs^T)^T   (rows pass of forward DFT, stored transposed)
__global__ void flg_fftrow_kernel(const _Float16* __restrict__ xh,
                                  const _Float16* __restrict__ Fsr,
                                  const _Float16* __restrict__ Fsi,
                                  _Float16* __restrict__ Trt, _Float16* __restrict__ Tit) {
  int wid = blockIdx.x * 8 + (threadIdx.x >> 5);
  int lane = threadIdx.x & 31;
  int img = wid / 144;
  int tt = wid % 144;
  int m0 = (tt / 12) * 16, n0 = (tt % 12) * 16;
  const _Float16* A = xh + (size_t)img * HW;
  v8f ar = zero8(), ai = zero8();
#pragma unroll 3
  for (int k0 = 0; k0 < WW; k0 += 32) {
    if (k0 + 32 < WW) {
      __builtin_prefetch(A + (size_t)(m0 + (lane & 15)) * WW + k0 + 32, 0, 1);
      __builtin_prefetch(Fsr + (size_t)(n0 + (lane & 15)) * WW + k0 + 32, 0, 1);
      __builtin_prefetch(Fsi + (size_t)(n0 + (lane & 15)) * WW + k0 + 32, 0, 1);
    }
    v16h a  = fragAh(A, WW, m0, k0, lane);
    v16h br = fragBTh(Fsr, WW, k0, n0, lane, false);  // B[k=w][n=v] = Fs[v,w]
    v16h bi = fragBTh(Fsi, WW, k0, n0, lane, false);
    ar = wmma16(a, br, ar);
    ai = wmma16(a, bi, ai);
  }
  storeTileHT(Trt + (size_t)img * HW, WW, m0, n0, lane, ar);
  storeTileHT(Tit + (size_t)img * HW, WW, m0, n0, lane, ai);
}

// X[perm(img)] = Fs * T[img]  (complex GEMM; batch/channel fftshift rolls
// folded into the destination slot). T read from transposed f16 storage.
__global__ void flg_fftcol_kernel(const _Float16* __restrict__ Fsr,
                                  const _Float16* __restrict__ Fsi,
                                  const _Float16* __restrict__ Trt,
                                  const _Float16* __restrict__ Tit,
                                  float* __restrict__ Xr, float* __restrict__ Xi) {
  int wid = blockIdx.x * 8 + (threadIdx.x >> 5);
  int lane = threadIdx.x & 31;
  int img = wid / 144;
  int tt = wid % 144;
  int m0 = (tt / 12) * 16, n0 = (tt % 12) * 16;
  int bs = img / CINC, cs = img % CINC;
  int dimg = ((bs + 2) & 3) * CINC + ((cs + 2) % 3);
  const _Float16* Br = Trt + (size_t)img * HW;   // [v][h] transposed
  const _Float16* Bi = Tit + (size_t)img * HW;
  v8f cr = zero8(), ci = zero8();
#pragma unroll 2
  for (int k0 = 0; k0 < HH; k0 += 32) {
    if (k0 + 32 < HH) {
      __builtin_prefetch(Br + (size_t)(n0 + (lane & 15)) * HH + k0 + 32, 0, 1);
      __builtin_prefetch(Bi + (size_t)(n0 + (lane & 15)) * HH + k0 + 32, 0, 1);
    }
    v16h arf = fragAh(Fsr, HH, m0, k0, lane);
    v16h aif = fragAh(Fsi, HH, m0, k0, lane);
    v16h brf = fragBTh(Br, HH, k0, n0, lane, false); // B[k=h][n=v] = T[h,v]
    v16h bif = fragBTh(Bi, HH, k0, n0, lane, false);
    v16h bin = fragBTh(Bi, HH, k0, n0, lane, true);
    cr = wmma16(arf, brf, cr);
    cr = wmma16(aif, bin, cr);
    ci = wmma16(arf, bif, ci);
    ci = wmma16(aif, brf, ci);
  }
  storeTile(Xr + (size_t)dimg * HW, WW, m0, n0, lane, cr);
  storeTile(Xi + (size_t)dimg * HW, WW, m0, n0, lane, ci);
}

// h1[pix][64] = relu(W1 * |X| + b1) -> f16, one thread per (b,pixel)
__global__ void flg_h1_kernel(const float* __restrict__ Xr,
                              const float* __restrict__ Xi,
                              const float* __restrict__ w1,
                              const float* __restrict__ b1,
                              _Float16* __restrict__ h1h) {
  int p = blockIdx.x * 256 + threadIdx.x;
  if (p >= NPIX) return;
  int b = p / HW, hw = p % HW;
  float mag[CINC];
#pragma unroll
  for (int i = 0; i < CINC; ++i) {
    float re = Xr[(size_t)(b * CINC + i) * HW + hw];
    float im = Xi[(size_t)(b * CINC + i) * HW + hw];
    mag[i] = sqrtf(re * re + im * im);
  }
  _Float16* out = h1h + (size_t)p * HID;
#pragma unroll 4
  for (int o = 0; o < HID; ++o) {
    float s = b1[o];
#pragma unroll
    for (int i = 0; i < CINC; ++i) s += w1[o * CINC + i] * mag[i];
    out[o] = (_Float16)fmaxf(s, 0.0f);
  }
}

// h2 = h1 * W2^T + b2 via WMMA; only per-channel BN statistics retained.
__global__ void flg_h2stats_kernel(const _Float16* __restrict__ h1h,
                                   const _Float16* __restrict__ w2h,
                                   const float* __restrict__ b2,
                                   float* __restrict__ sums,
                                   float* __restrict__ sumsq) {
  int wid = blockIdx.x * 8 + (threadIdx.x >> 5);
  int lane = threadIdx.x & 31;
  int rt = wid / 12, ct = wid % 12;
  int m0 = rt * 16, n0 = ct * 16;
  v8f acc = zero8();
#pragma unroll
  for (int k0 = 0; k0 < HID; k0 += 32) {
    v16h a = fragAh(h1h, HID, m0, k0, lane);
    v16h b = fragBTh(w2h, HID, k0, n0, lane, false); // B[k=hid][n=ch]=w2[ch,hid]
    acc = wmma16(a, b, acc);
  }
  int col = n0 + (lane & 15);
  float bias = b2[col];
  float s = 0.0f, q = 0.0f;
#pragma unroll
  for (int j = 0; j < 8; ++j) {
    float v = acc[j] + bias;
    s += v;
    q += v * v;
  }
  s += __shfl_xor(s, 16, 32);   // lanes L and L^16 hold the same column
  q += __shfl_xor(q, 16, 32);
  if (lane < 16) {
    atomicAdd(&sums[col], s);
    atomicAdd(&sumsq[col], q);
  }
}

__global__ void flg_bnfinal_kernel(const float* __restrict__ sums,
                                   const float* __restrict__ sumsq,
                                   const float* __restrict__ bnw,
                                   const float* __restrict__ bnb,
                                   float* __restrict__ scale,
                                   float* __restrict__ shift) {
  int c = threadIdx.x;
  if (c >= NCH) return;
  const float N = (float)NPIX;
  float mu  = sums[c] / N;
  float var = sumsq[c] / N - mu * mu;
  float rs  = rsqrtf(var + 1e-5f);
  scale[c] = bnw[c] * rs;
  shift[c] = bnb[c] - mu * bnw[c] * rs;
}

// One wave per pixel: recompute h2, BN-normalize, softmax(192), evaluate the
// log-Gabor bank g[s,o,i] and accumulate A_tot = sum_{b,s,o,i} attn*g*X.
__global__ void flg_attend_kernel(const _Float16* __restrict__ h1h,
                                  const float* __restrict__ w2,
                                  const float* __restrict__ b2,
                                  const float* __restrict__ scale,
                                  const float* __restrict__ shift,
                                  const float* __restrict__ Xr,
                                  const float* __restrict__ Xi,
                                  const float* __restrict__ freq,
                                  const float* __restrict__ theta,
                                  const float* __restrict__ sigma,
                                  const float* __restrict__ f0,
                                  const float* __restrict__ th0,
                                  float* __restrict__ Ar,
                                  float* __restrict__ Ai) {
  int wid = blockIdx.x * 8 + (threadIdx.x >> 5);   // pixel index (0..HW-1)
  int lane = threadIdx.x & 31;
  int hh = wid / WW, ww = wid % WW;
  float yv = -1.0f + 2.0f * (float)hh / 191.0f;    // linspace(-1,1,192)
  float xv = -1.0f + 2.0f * (float)ww / 191.0f;
  float r = sqrtf(xv * xv + yv * yv + 1e-6f);
  float phi = atan2f(yv, xv);
  float logr = logf(r);

  // h2 (normalized) for this pixel: lane handles channels lane + 32*t
  float h2v[BB][6];
#pragma unroll
  for (int b = 0; b < BB; ++b) {
    const _Float16* h1p = h1h + (size_t)(b * HW + wid) * HID;
#pragma unroll
    for (int t = 0; t < 6; ++t) {
      int ch = lane + 32 * t;
      float s = b2[ch];
      const float* wrow = w2 + (size_t)ch * HID;
#pragma unroll 8
      for (int k = 0; k < HID; ++k) s += wrow[k] * (float)h1p[k];
      h2v[b][t] = s * scale[ch] + shift[ch];
    }
  }

  // softmax over the 192 channels, per batch
  float attnv[BB][6];
#pragma unroll
  for (int b = 0; b < BB; ++b) {
    float m = h2v[b][0];
#pragma unroll
    for (int t = 1; t < 6; ++t) m = fmaxf(m, h2v[b][t]);
    m = waveMax(m);
    float sum = 0.0f;
#pragma unroll
    for (int t = 0; t < 6; ++t) {
      float e = expf(h2v[b][t] - m);
      attnv[b][t] = e;
      sum += e;
    }
    sum = waveSum(sum);
    float inv = 1.0f / sum;
#pragma unroll
    for (int t = 0; t < 6; ++t) attnv[b][t] *= inv;
  }

  // log-Gabor bank & accumulation of Wri[b][i] = sum_{s,o} attn * g
  float W[BB][CINC];
#pragma unroll
  for (int b = 0; b < BB; ++b)
#pragma unroll
    for (int i = 0; i < CINC; ++i) W[b][i] = 0.0f;

#pragma unroll
  for (int t = 0; t < 6; ++t) {
    int ch = lane + 32 * t;
    int sc = ch >> 6;     // scale index (reshape [NSC, COUT])
    int oc = ch & 63;     // out-channel index
#pragma unroll
    for (int i = 0; i < CINC; ++i) {
      int pidx = (sc * COUTC + oc) * CINC + i;
      float fr  = freq[pidx];
      float th  = theta[pidx];
      float sg  = sigma[pidx];
      float f0v = f0[pidx];
      float t0  = th0[pidx];
      float lg = logr - logf(f0v);
      float ls = logf(sg / f0v);
      float dph = phi - th;
      float g = expf(-(lg * lg) / (2.0f * ls * ls)) *
                expf(-(dph * dph) / (2.0f * t0 * t0)) *
                cosf(fr * r) / (2.0f * PI_F * sg * sg);
#pragma unroll
      for (int b = 0; b < BB; ++b) W[b][i] += attnv[b][t] * g;
    }
  }

#pragma unroll
  for (int b = 0; b < BB; ++b)
#pragma unroll
    for (int i = 0; i < CINC; ++i) W[b][i] = waveSum(W[b][i]);

  if (lane == 0) {
    float ar = 0.0f, ai = 0.0f;
#pragma unroll
    for (int b = 0; b < BB; ++b)
#pragma unroll
      for (int i = 0; i < CINC; ++i) {
        float wv = W[b][i];
        ar += wv * Xr[(size_t)(b * CINC + i) * HW + wid];
        ai += wv * Xi[(size_t)(b * CINC + i) * HW + wid];
      }
    Ar[wid] = ar;
    Ai[wid] = ai;
  }
}

// Q^T[o][v][u] = A_tot[u,v] * Msum[(o+32)%64][u,v] / 192^2  (f16, transposed,
// ifftshift o-roll and full inverse-DFT normalization folded here)
__global__ void flg_buildq_kernel(const float* __restrict__ Ar,
                                  const float* __restrict__ Ai,
                                  const int* __restrict__ idxlo,
                                  const int* __restrict__ idxhi,
                                  _Float16* __restrict__ Qrt, _Float16* __restrict__ Qit) {
  int t = blockIdx.x * 256 + threadIdx.x;   // o*HW + v*192 + u
  int o = t / HW, vu = t % HW;
  int v = vu / WW, u = vu % WW;             // consecutive t -> consecutive u
  int os = (o + 32) & 63;
  float ms = 0.0f;
#pragma unroll
  for (int i = 0; i < CINC; ++i) {
    int lo = idxlo[os * CINC + i];
    int hi = idxhi[os * CINC + i];
    bool inb = (u >= lo) && (u < hi) && (v >= lo) && (v < hi);
    ms += inb ? 1.0f : 0.0f;
  }
  ms *= (1.0f / (192.0f * 192.0f));
  int hw = u * WW + v;                       // A_tot is [h=u][w=v]
  Qrt[t] = (_Float16)(Ar[hw] * ms);
  Qit[t] = (_Float16)(Ai[hw] * ms);
}

// R[o] = IFs * Q[o]  (complex GEMM; Q from transposed f16 storage)
__global__ void flg_ifftrow_kernel(const _Float16* __restrict__ IFsr,
                                   const _Float16* __restrict__ IFsi,
                                   const _Float16* __restrict__ Qrt,
                                   const _Float16* __restrict__ Qit,
                                   _Float16* __restrict__ Rr, _Float16* __restrict__ Ri) {
  int wid = blockIdx.x * 8 + (threadIdx.x >> 5);
  int lane = threadIdx.x & 31;
  int o = wid / 144;
  int tt = wid % 144;
  int m0 = (tt / 12) * 16, n0 = (tt % 12) * 16;
  const _Float16* Br = Qrt + (size_t)o * HW;  // [v][u] transposed
  const _Float16* Bi = Qit + (size_t)o * HW;
  v8f cr = zero8(), ci = zero8();
#pragma unroll 2
  for (int k0 = 0; k0 < HH; k0 += 32) {
    if (k0 + 32 < HH) {
      __builtin_prefetch(Br + (size_t)(n0 + (lane & 15)) * HH + k0 + 32, 0, 1);
      __builtin_prefetch(Bi + (size_t)(n0 + (lane & 15)) * HH + k0 + 32, 0, 1);
    }
    v16h arf = fragAh(IFsr, HH, m0, k0, lane);
    v16h aif = fragAh(IFsi, HH, m0, k0, lane);
    v16h brf = fragBTh(Br, HH, k0, n0, lane, false); // B[k=u][n=v] = Q[u,v]
    v16h bif = fragBTh(Bi, HH, k0, n0, lane, false);
    v16h bin = fragBTh(Bi, HH, k0, n0, lane, true);
    cr = wmma16(arf, brf, cr);
    cr = wmma16(aif, bin, cr);
    ci = wmma16(arf, bif, ci);
    ci = wmma16(aif, brf, ci);
  }
  storeTileH(Rr + (size_t)o * HW, WW, m0, n0, lane, cr);
  storeTileH(Ri + (size_t)o * HW, WW, m0, n0, lane, ci);
}

// xf[o] = Re{ R[o] * IFs^T } = Rr*IFsr^T - Ri*IFsi^T
__global__ void flg_ifftcol_kernel(const _Float16* __restrict__ IFsr,
                                   const _Float16* __restrict__ IFsi,
                                   const _Float16* __restrict__ Rr,
                                   const _Float16* __restrict__ Ri,
                                   float* __restrict__ xf) {
  int wid = blockIdx.x * 8 + (threadIdx.x >> 5);
  int lane = threadIdx.x & 31;
  int o = wid / 144;
  int tt = wid % 144;
  int m0 = (tt / 12) * 16, n0 = (tt % 12) * 16;
  const _Float16* Aru = Rr + (size_t)o * HW;
  const _Float16* Aiu = Ri + (size_t)o * HW;
  v8f acc = zero8();
#pragma unroll 3
  for (int k0 = 0; k0 < WW; k0 += 32) {
    if (k0 + 32 < WW) {
      __builtin_prefetch(Aru + (size_t)(m0 + (lane & 15)) * WW + k0 + 32, 0, 1);
      __builtin_prefetch(Aiu + (size_t)(m0 + (lane & 15)) * WW + k0 + 32, 0, 1);
    }
    v16h ar = fragAh(Aru, WW, m0, k0, lane);
    v16h ai = fragAh(Aiu, WW, m0, k0, lane);
    v16h br = fragBTh(IFsr, WW, k0, n0, lane, false); // B[k=v][n] = IFs[n,v]
    v16h bi = fragBTh(IFsi, WW, k0, n0, lane, true);
    acc = wmma16(ar, br, acc);
    acc = wmma16(ai, bi, acc);
  }
  storeTile(xf + (size_t)o * HW, WW, m0, n0, lane, acc);
}

// out = mix*xf + (1-mix)*conv3x3(x)
__global__ void flg_convmix_kernel(const float* __restrict__ x,
                                   const float* __restrict__ cw,
                                   const float* __restrict__ mixp,
                                   const float* __restrict__ xf,
                                   float* __restrict__ out) {
  int t = blockIdx.x * 256 + threadIdx.x;   // < B*COUT*HW = 9437184
  int w = t % WW;
  int h = (t / WW) % HH;
  int o = (t / HW) % COUTC;
  int b = t / (HW * COUTC);
  float acc = 0.0f;
#pragma unroll
  for (int i = 0; i < CINC; ++i) {
    const float* xp = x + (size_t)(b * CINC + i) * HW;
    const float* wp = cw + (size_t)(o * CINC + i) * 9;
#pragma unroll
    for (int kh = 0; kh < 3; ++kh) {
      int ih = h + kh - 1;
      if (ih < 0 || ih >= HH) continue;
#pragma unroll
      for (int kw = 0; kw < 3; ++kw) {
        int iw = w + kw - 1;
        if (iw < 0 || iw >= WW) continue;
        acc += wp[kh * 3 + kw] * xp[ih * WW + iw];
      }
    }
  }
  float mix = mixp[0];
  out[t] = mix * xf[(size_t)o * HW + h * WW + w] + (1.0f - mix) * acc;
}

// ------------------------------ launcher -----------------------------------

extern "C" void kernel_launch(void* const* d_in, const int* in_sizes, int n_in,
                              void* d_out, int out_size, void* d_ws, size_t ws_size,
                              hipStream_t stream) {
  (void)in_sizes; (void)n_in; (void)out_size; (void)ws_size;

  const float* x      = (const float*)d_in[0];
  const float* freq   = (const float*)d_in[1];
  const float* theta  = (const float*)d_in[2];
  const float* sigma  = (const float*)d_in[3];
  const float* f0     = (const float*)d_in[4];
  const float* theta0 = (const float*)d_in[5];
  const float* aw1    = (const float*)d_in[6];
  const float* ab1    = (const float*)d_in[7];
  const float* aw2    = (const float*)d_in[8];
  const float* ab2    = (const float*)d_in[9];
  const float* bnw    = (const float*)d_in[10];
  const float* bnb    = (const float*)d_in[11];
  const float* mixp   = (const float*)d_in[12];
  const float* convw  = (const float*)d_in[13];
  const float* fbs    = (const float*)d_in[14];
  float* out = (float*)d_out;

  char* wsb = (char*)d_ws;
  size_t off = 0;
  auto alloc = [&](size_t bytes) -> char* {
    char* p = wsb + off;
    off += (bytes + 255) & ~(size_t)255;
    return p;
  };

  _Float16* Fsr  = (_Float16*)alloc(HW * 2);
  _Float16* Fsi  = (_Float16*)alloc(HW * 2);
  _Float16* IFsr = (_Float16*)alloc(HW * 2);
  _Float16* IFsi = (_Float16*)alloc(HW * 2);
  _Float16* xh   = (_Float16*)alloc((size_t)BB * CINC * HW * 2);
  _Float16* w2h  = (_Float16*)alloc((size_t)NCH * HID * 2);
  _Float16* Trt  = (_Float16*)alloc((size_t)BB * CINC * HW * 2);
  _Float16* Tit  = (_Float16*)alloc((size_t)BB * CINC * HW * 2);
  float*    Xr   = (float*)alloc((size_t)BB * CINC * HW * 4);
  float*    Xi   = (float*)alloc((size_t)BB * CINC * HW * 4);
  _Float16* h1h  = (_Float16*)alloc((size_t)NPIX * HID * 2);
  float*    stats = (float*)alloc(384 * 4);   // sums[192] | sumsq[192]
  float*    sums  = stats;
  float*    sumsq = stats + 192;
  float*    scale = (float*)alloc(192 * 4);
  float*    shift = (float*)alloc(192 * 4);
  int*      idxlo = (int*)alloc(192 * 4);
  int*      idxhi = (int*)alloc(192 * 4);
  float*    Ar   = (float*)alloc(HW * 4);
  float*    Ai   = (float*)alloc(HW * 4);
  _Float16* Qrt  = (_Float16*)alloc((size_t)COUTC * HW * 2);
  _Float16* Qit  = (_Float16*)alloc((size_t)COUTC * HW * 2);
  _Float16* Rr   = (_Float16*)alloc((size_t)COUTC * HW * 2);
  _Float16* Ri   = (_Float16*)alloc((size_t)COUTC * HW * 2);
  float*    xf   = (float*)alloc((size_t)COUTC * HW * 4);

  // 1. twiddle tables (f16), x/w2 -> f16, band indices, zero BN stats
  flg_tables_kernel<<<144, 256, 0, stream>>>(fbs, x, aw2, Fsr, Fsi, IFsr, IFsi,
                                             xh, w2h, idxlo, idxhi, stats);
  // 2-3. forward DFT (12 images): 144 tiles each, 8 waves/block
  flg_fftrow_kernel<<<(12 * 144) / 8, 256, 0, stream>>>(xh, Fsr, Fsi, Trt, Tit);
  flg_fftcol_kernel<<<(12 * 144) / 8, 256, 0, stream>>>(Fsr, Fsi, Trt, Tit, Xr, Xi);
  // 4. attention hidden layer
  flg_h1_kernel<<<NPIX / 256, 256, 0, stream>>>(Xr, Xi, aw1, ab1, h1h);
  // 5. h2 GEMM (WMMA) -> BN statistics only
  flg_h2stats_kernel<<<((NPIX / 16) * 12) / 8, 256, 0, stream>>>(h1h, w2h, ab2,
                                                                 sums, sumsq);
  // 6. finalize BN affine
  flg_bnfinal_kernel<<<1, 192, 0, stream>>>(sums, sumsq, bnw, bnb, scale, shift);
  // 7. fused softmax + log-Gabor + complex accumulate (one wave / pixel)
  flg_attend_kernel<<<HW / 8, 256, 0, stream>>>(h1h, aw2, ab2, scale, shift,
                                                Xr, Xi, freq, theta, sigma,
                                                f0, theta0, Ar, Ai);
  // 8. Q[o] = A_tot * band-mask (o-roll + 1/192^2 folded), f16 transposed
  flg_buildq_kernel<<<(COUTC * HW) / 256, 256, 0, stream>>>(Ar, Ai, idxlo, idxhi,
                                                            Qrt, Qit);
  // 9-10. inverse DFT over 64 batch-invariant channels
  flg_ifftrow_kernel<<<(COUTC * 144) / 8, 256, 0, stream>>>(IFsr, IFsi, Qrt, Qit,
                                                            Rr, Ri);
  flg_ifftcol_kernel<<<(COUTC * 144) / 8, 256, 0, stream>>>(IFsr, IFsi, Rr, Ri, xf);
  // 11. 3x3 conv + mix
  flg_convmix_kernel<<<(BB * COUTC * HW) / 256, 256, 0, stream>>>(x, convw, mixp,
                                                                  xf, out);
}
